// SelfAttention_14285061226820
// MI455X (gfx1250) — compile-verified, matrix operand order checked by reading, not run on hardware
//
#include <hip/hip_runtime.h>
#include <hip/hip_bf16.h>

// ---------------------------------------------------------------------------
// Fused causal single-head attention for MI455X (gfx1250, wave32, WMMA).
//   x:[4,4096,1024] f32, Wq/Wk/Wv:[1024,64] f32  ->  out:[4,4096,64] f32
// Pipeline:
//   k0: Wq/Wk/Wv f32 -> WT bf16 [3][64][1024] (transposed for contiguous B-frags)
//   k1: QKV projection with v_wmma_f32_16x16x32_bf16; Q pre-scaled by 1/32,
//       V stored transposed VT[b][64][T] (bf16) for contiguous P@V B-frags.
//   k2: flash attention: double-buffered async-to-LDS staging of K/V tiles
//       (global_load_async_to_lds_b128 + s_wait_asynccnt), WMMA S=QK^T and
//       O+=PV, online softmax with wave32 shfl reductions, LDS P bounce.
// Workspace: 3*64*1024*2 + 3*(16384*64*2) = 6,684,672 bytes.
// ---------------------------------------------------------------------------

#define EMBED 1024
#define HDIM  64
#define BATCH 4
#define TSEQ  4096
#define BT    (BATCH * TSEQ)

typedef __attribute__((ext_vector_type(16))) __bf16       v16bf;
typedef __attribute__((ext_vector_type(8)))  float        v8f;
typedef __attribute__((ext_vector_type(8)))  unsigned int v8u;

__device__ __forceinline__ unsigned short f2bf(float f) {
    unsigned int u = __builtin_bit_cast(unsigned int, f);
    unsigned int r = u + 0x7FFFu + ((u >> 16) & 1u);   // round-to-nearest-even
    return (unsigned short)(r >> 16);
}
__device__ __forceinline__ unsigned int pack2(float lo, float hi) {
    return (unsigned int)f2bf(lo) | ((unsigned int)f2bf(hi) << 16);
}
__device__ __forceinline__ v8f wmma_bf16(v16bf a, v16bf b, v8f c) {
    return __builtin_amdgcn_wmma_f32_16x16x32_bf16(false, a, false, b,
                                                   (short)0, c, false, false);
}
// K-index of VGPR pair j for the 16-bit 16x32 A layout (ISA 7.12.2); B mirrors it.
__device__ __forceinline__ int frag_bk(int j, int half) {
    return ((j < 4) ? 2 * j : 2 * j + 8) + 8 * half;
}
// Async DMA: 16 bytes per lane, global -> LDS (ASYNCcnt).  VDST = LDS byte
// offset VGPR, VADDR = 64-bit global address (ISA 15.18.3 op 98).
__device__ __forceinline__ void async_cp16(const void* gaddr, void* lds_ptr) {
    unsigned lds_off = (unsigned)(unsigned long long)lds_ptr;   // low 32b = LDS offset
    asm volatile("global_load_async_to_lds_b128 %0, %1, off"
                 :: "v"(lds_off), "v"((unsigned long long)gaddr) : "memory");
}
__device__ __forceinline__ void wait_async0() {
    asm volatile("s_wait_asynccnt 0x0" ::: "memory");
}

// ---------------- kernel 0: weight convert + transpose ---------------------
__global__ void wconv_kernel(const float* __restrict__ Wq,
                             const float* __restrict__ Wk,
                             const float* __restrict__ Wv,
                             unsigned short* __restrict__ WT) {
    int idx = blockIdx.x * 256 + threadIdx.x;          // 3*64*1024 total
    if (idx >= 3 * HDIM * EMBED) return;
    int mat = idx >> 16;
    int rem = idx & 0xFFFF;
    int n = rem >> 10;          // output column 0..63
    int k = rem & 1023;         // embed index
    const float* W = (mat == 0) ? Wq : ((mat == 1) ? Wk : Wv);
    WT[idx] = f2bf(W[k * HDIM + n]);                   // WT[mat][n][k]
}

// ---------------- kernel 1: QKV projection (WMMA) --------------------------
__global__ void __launch_bounds__(256)
qkv_kernel(const float* __restrict__ x,
           const unsigned short* __restrict__ WT,   // [3][64][1024] bf16
           unsigned short* __restrict__ Qb,         // [BT][64] bf16 (pre-scaled)
           unsigned short* __restrict__ Kb,         // [BT][64] bf16
           unsigned short* __restrict__ VT) {       // [B][64][T]  bf16
    const int lane = threadIdx.x & 31;
    const int wave = threadIdx.x >> 5;
    const int r    = lane & 15;
    const int half = lane >> 4;
    const int rowbase = blockIdx.x * 128 + wave * 16;   // 16-row band per wave

    v8f qa[4], ka[4], va[4];
#pragma unroll
    for (int nt = 0; nt < 4; ++nt) { qa[nt] = {}; ka[nt] = {}; va[nt] = {}; }

    const unsigned short* WTq = WT;
    const unsigned short* WTk = WT + HDIM * EMBED;
    const unsigned short* WTv = WT + 2 * HDIM * EMBED;

    for (int kk = 0; kk < EMBED; kk += 32) {
        // A fragment: 16 rows of x, K = kk..kk+31, f32 -> bf16
        v8u au;
        const float* xrow = x + (size_t)(rowbase + r) * EMBED + kk;
#pragma unroll
        for (int j = 0; j < 8; ++j) {
            int bk = frag_bk(j, half);
            float2 f = *(const float2*)(xrow + bk);
            au[j] = pack2(f.x, f.y);
        }
        v16bf a = __builtin_bit_cast(v16bf, au);

#pragma unroll
        for (int nt = 0; nt < 4; ++nt) {
            int c = 16 * nt + r;                       // output column
            size_t wo = (size_t)c * EMBED + kk;
            v8u bq, bk_, bv;
#pragma unroll
            for (int j = 0; j < 8; ++j) {
                int bk = frag_bk(j, half);
                bq[j]  = *(const unsigned int*)(WTq + wo + bk);
                bk_[j] = *(const unsigned int*)(WTk + wo + bk);
                bv[j]  = *(const unsigned int*)(WTv + wo + bk);
            }
            qa[nt] = wmma_bf16(a, __builtin_bit_cast(v16bf, bq),  qa[nt]);
            ka[nt] = wmma_bf16(a, __builtin_bit_cast(v16bf, bk_), ka[nt]);
            va[nt] = wmma_bf16(a, __builtin_bit_cast(v16bf, bv),  va[nt]);
        }
    }

    // Epilogue: D layout is (m = j + 8*half, n = 16*nt + (lane&15))
    const int   bidx = rowbase >> 12;                  // / TSEQ
    const int   t0   = rowbase & (TSEQ - 1);
    const float qscale = 0.03125f;                     // 1/sqrt(1024)
#pragma unroll
    for (int nt = 0; nt < 4; ++nt) {
#pragma unroll
        for (int j = 0; j < 8; ++j) {
            int m = j + 8 * half;
            int n = 16 * nt + r;
            size_t row = (size_t)(rowbase + m);
            Qb[row * HDIM + n] = f2bf(qa[nt][j] * qscale);
            Kb[row * HDIM + n] = f2bf(ka[nt][j]);
            VT[((size_t)bidx * HDIM + n) * TSEQ + (t0 + m)] = f2bf(va[nt][j]);
        }
    }
}

// ---------------- kernel 2: causal flash attention -------------------------
__global__ void __launch_bounds__(256)
flash_kernel(const unsigned short* __restrict__ Qb,
             const unsigned short* __restrict__ Kb,
             const unsigned short* __restrict__ VT,
             float* __restrict__ out) {
    __shared__ unsigned short Ks[2][32 * HDIM];        // K tile  [32 keys][64]
    __shared__ unsigned short Vs[2][HDIM * 32];        // V^T tile [64][32 keys]
    __shared__ unsigned short Pbuf[8][16 * 32];        // per-wave P bounce

    const int tid  = threadIdx.x;
    const int lane = tid & 31;
    const int wave = tid >> 5;
    const int r    = lane & 15;
    const int half = lane >> 4;
    const int b    = blockIdx.y;
    const int qb   = blockIdx.x * 128;                 // block q rows [qb, qb+128)
    const int qwave = qb + wave * 16;                  // this wave's q rows
    const size_t qrow0 = (size_t)b * TSEQ + qwave;
    const size_t bT    = (size_t)b * TSEQ;

    float m_run[8], l_run[8];
    v8f oacc[4];
#pragma unroll
    for (int j = 0; j < 8; ++j) { m_run[j] = -3.0e38f; l_run[j] = 0.0f; }
#pragma unroll
    for (int nt = 0; nt < 4; ++nt) oacc[nt] = {};

    // Q A-fragments (rows fixed for the whole loop), hh in {0,32}
    v16bf aq[2];
#pragma unroll
    for (int h2 = 0; h2 < 2; ++h2) {
        v8u u;
        const unsigned short* qp = Qb + (qrow0 + r) * HDIM + h2 * 32;
#pragma unroll
        for (int j = 0; j < 8; ++j) u[j] = *(const unsigned int*)(qp + frag_bk(j, half));
        aq[h2] = __builtin_bit_cast(v16bf, u);
    }

    unsigned short* myP = &Pbuf[wave][0];

    // Cooperative async stage of one 32-key tile (8 KB): 16 B per thread x2.
    const int krow = tid >> 3, kseg = tid & 7;         // K: [32][8 x 16B]
    const int vrow = tid >> 2, vseg = tid & 3;         // V: [64][4 x 16B]
    auto stage = [&](int bufi, int kk0) {
        async_cp16(Kb + (bT + kk0 + krow) * HDIM + kseg * 8,
                   &Ks[bufi][krow * HDIM + kseg * 8]);
        async_cp16(VT + ((size_t)b * HDIM + vrow) * TSEQ + kk0 + vseg * 8,
                   &Vs[bufi][vrow * 32 + vseg * 8]);
    };

    const int trips = (qb + 128) >> 5;                 // block-uniform
    stage(0, 0);
    for (int it = 0; it < trips; ++it) {
        const int kk = it << 5;
        const int cb = it & 1;
        wait_async0();                                 // my staging complete
        __syncthreads();                               // tile visible block-wide
        if (it + 1 < trips) stage(cb ^ 1, kk + 32);    // overlap next DMA

        if (kk <= qwave + 15) {                        // causal: skip above diag
            // ---- S = Q K^T for 32 keys (two 16-wide D tiles) ----
            v8f s[2]; s[0] = {}; s[1] = {};
#pragma unroll
            for (int st = 0; st < 2; ++st) {
                const unsigned short* kp = &Ks[cb][(16 * st + r) * HDIM];
#pragma unroll
                for (int h2 = 0; h2 < 2; ++h2) {
                    v8u u;
#pragma unroll
                    for (int j = 0; j < 8; ++j)
                        u[j] = *(const unsigned int*)(kp + h2 * 32 + frag_bk(j, half));
                    s[st] = wmma_bf16(aq[h2], __builtin_bit_cast(v16bf, u), s[st]);
                }
                if (kk + 16 * st + 15 > qwave) {       // diagonal tile: mask
                    int key = kk + 16 * st + r;
#pragma unroll
                    for (int j = 0; j < 8; ++j)
                        if (key > qwave + j + 8 * half) s[st][j] = -__builtin_inff();
                }
            }

            // ---- online softmax: row reductions across lanes of each half ----
            float alpha[8];
#pragma unroll
            for (int j = 0; j < 8; ++j) {
                float v = fmaxf(s[0][j], s[1][j]);
                v = fmaxf(v, __shfl_xor(v, 1));
                v = fmaxf(v, __shfl_xor(v, 2));
                v = fmaxf(v, __shfl_xor(v, 4));
                v = fmaxf(v, __shfl_xor(v, 8));
                float mnew = fmaxf(m_run[j], v);
                float al   = __expf(m_run[j] - mnew);
                float p0   = __expf(s[0][j] - mnew);
                float p1   = __expf(s[1][j] - mnew);
                float rs   = p0 + p1;
                rs += __shfl_xor(rs, 1);
                rs += __shfl_xor(rs, 2);
                rs += __shfl_xor(rs, 4);
                rs += __shfl_xor(rs, 8);
                m_run[j] = mnew;
                l_run[j] = l_run[j] * al + rs;
                alpha[j] = al;
                // P -> LDS in row-major [16][32] bf16 (D layout scatter)
                myP[(j + 8 * half) * 32 + r]      = f2bf(p0);
                myP[(j + 8 * half) * 32 + 16 + r] = f2bf(p1);
            }
#pragma unroll
            for (int nt = 0; nt < 4; ++nt)
#pragma unroll
                for (int j = 0; j < 8; ++j) oacc[nt][j] *= alpha[j];

            asm volatile("s_wait_dscnt 0x0" ::: "memory"); // P store->load fence

            // ---- P A-fragment (16x32) from LDS ----
            v8u pu;
#pragma unroll
            for (int j = 0; j < 8; ++j)
                pu[j] = *(const unsigned int*)(myP + r * 32 + frag_bk(j, half));
            v16bf pa = __builtin_bit_cast(v16bf, pu);

            // ---- O += P @ V (V^T tile: contiguous along key dim) ----
#pragma unroll
            for (int nt = 0; nt < 4; ++nt) {
                const unsigned short* vp = &Vs[cb][(16 * nt + r) * 32];
                v8u vu;
#pragma unroll
                for (int j = 0; j < 8; ++j)
                    vu[j] = *(const unsigned int*)(vp + frag_bk(j, half));
                oacc[nt] = wmma_bf16(pa, __builtin_bit_cast(v16bf, vu), oacc[nt]);
            }
        }
    }

    // ---- normalize and store fp32 output ----
    float inv[8];
#pragma unroll
    for (int j = 0; j < 8; ++j) inv[j] = 1.0f / l_run[j];
#pragma unroll
    for (int nt = 0; nt < 4; ++nt)
#pragma unroll
        for (int j = 0; j < 8; ++j) {
            int m = j + 8 * half;
            out[(qrow0 + m) * HDIM + 16 * nt + r] = oacc[nt][j] * inv[j];
        }
}

// ---------------------------------------------------------------------------
extern "C" void kernel_launch(void* const* d_in, const int* in_sizes, int n_in,
                              void* d_out, int out_size, void* d_ws, size_t ws_size,
                              hipStream_t stream) {
    const float* x  = (const float*)d_in[0];
    const float* Wq = (const float*)d_in[1];
    const float* Wk = (const float*)d_in[2];
    const float* Wv = (const float*)d_in[3];
    float* out = (float*)d_out;

    char* ws = (char*)d_ws;
    unsigned short* WT = (unsigned short*)ws;                         // 393,216 B
    unsigned short* Qb = (unsigned short*)(ws + 3 * HDIM * EMBED * 2);
    unsigned short* Kb = Qb + (size_t)BT * HDIM;                      // +2 MB
    unsigned short* VT = Kb + (size_t)BT * HDIM;                      // +2 MB

    wconv_kernel<<<(3 * HDIM * EMBED + 255) / 256, 256, 0, stream>>>(Wq, Wk, Wv, WT);
    qkv_kernel<<<BT / 128, 256, 0, stream>>>(x, WT, Qb, Kb, VT);
    flash_kernel<<<dim3(TSEQ / 128, BATCH), 256, 0, stream>>>(Qb, Kb, VT, out);
}
